// PRMPConvLayer_1099511628138
// MI455X (gfx1250) — matrix operand order repaired and checked by reading, not compiled, not run on hardware
//
#include <hip/hip_runtime.h>
#include <hip/hip_bf16.h>

// ---------------------------------------------------------------------------
// CDNA5 (gfx1250) fused PRMP GNN layer.
// All edge-level GEMMs are algebraically pushed to node level (scatter_mean is
// linear; predicted_child is constant per destination), and all chained linear
// maps are folded into precomputed weight products so only two fused node
// kernels carry the WMMA work.
// ---------------------------------------------------------------------------

typedef float v8f  __attribute__((ext_vector_type(8)));
typedef __bf16 v16bf __attribute__((ext_vector_type(16)));
typedef __bf16 v8bf  __attribute__((ext_vector_type(8)));

#define D_DIM 128

// ---------------------------------------------------------------------------
// Packed weight layout for B-fragments of v_wmma_f32_16x16x32_bf16.
// For W (K x N) row-major, fragment (kt, nt) for lane l holds 16 contiguous
// bf16: element j <-> W[kt*32 + (l>=16?16:0) + j][nt*16 + (l&15)].
// packed index = (((kt*NT)+nt)*32 + lane)*16 + j
// ---------------------------------------------------------------------------

// C[k][n] = alpha * sum_q A[k][q]*B[q][n]  +  addB * B[k][n], written packed.
// A == nullptr -> pure pack of B (use alpha=0, addB=1).
__global__ void fold_pack_kernel(const float* __restrict__ A,
                                 const float* __restrict__ B,
                                 float alpha, float addB,
                                 __bf16* __restrict__ outPk,
                                 int K, int Q, int N) {
    int t = blockIdx.x * blockDim.x + threadIdx.x;
    if (t >= K * N) return;
    int k = t / N, n = t - k * N;
    float acc = 0.0f;
    if (A) {
        float s = 0.0f;
        for (int q = 0; q < Q; ++q) s += A[k * Q + q] * B[q * N + n];
        acc = alpha * s;
    }
    if (addB != 0.0f) acc += addB * B[k * N + n];
    int kt = k >> 5, r = k & 31, kh = r >> 4, j = r & 15;
    int nt = n >> 4, lane = (kh << 4) | (n & 15);
    int NT = N >> 4;
    outPk[((((kt * NT) + nt) * 32 + lane) << 4) + j] = (__bf16)acc;
}

// Folded bias vectors (each length 128), one thread per output column.
__global__ void bias_fold_kernel(const float* __restrict__ b_self_c,
                                 const float* __restrict__ Wcomb_c,
                                 const float* __restrict__ b_comb_c,
                                 const float* __restrict__ b_msg_rev,
                                 const float* __restrict__ b_self_p,
                                 const float* __restrict__ Wcomb_p,
                                 const float* __restrict__ b_comb_p,
                                 const float* __restrict__ b_msg_fk,
                                 const float* __restrict__ b_pred2,
                                 float* __restrict__ beff_c,
                                 float* __restrict__ bfold_c,
                                 float* __restrict__ beff_p,
                                 float* __restrict__ bmask_p) {
    int n = threadIdx.x;  // 0..127
    float a0 = 0.f, a1 = 0.f, a2 = 0.f, a3 = 0.f;
    for (int q = 0; q < D_DIM; ++q) {
        float wca = Wcomb_c[q * D_DIM + n];
        float wcb = Wcomb_c[(q + D_DIM) * D_DIM + n];
        float wpa = Wcomb_p[q * D_DIM + n];
        float wpb = Wcomb_p[(q + D_DIM) * D_DIM + n];
        a0 += b_self_c[q] * wca;
        a1 += b_msg_rev[q] * wcb;
        a2 += b_self_p[q] * wpa;
        a3 += (b_msg_fk[q] - b_pred2[q]) * wpb;
    }
    beff_c[n]  = a0 + b_comb_c[n];
    bfold_c[n] = a1;
    beff_p[n]  = a2 + b_comb_p[n];
    bmask_p[n] = 0.5f * a3;
}

// ---------------------------------------------------------------------------
// Edge scatter: 32 threads per edge, 4 floats each, f32 atomics (L2-resident).
// ---------------------------------------------------------------------------
__global__ void scatter_kernel(const float* __restrict__ x,
                               const int* __restrict__ src,
                               const int* __restrict__ dst,
                               float* __restrict__ sum,
                               float* __restrict__ cnt, int nE) {
    int tid = blockIdx.x * blockDim.x + threadIdx.x;
    int e = tid >> 5;
    int c = tid & 31;
    if (e >= nE) return;
    int s = src[e], d = dst[e];
    const float4 v = *(const float4*)(x + (size_t)s * D_DIM + c * 4);
    float* o = sum + (size_t)d * D_DIM + c * 4;
    atomicAdd(o + 0, v.x);
    atomicAdd(o + 1, v.y);
    atomicAdd(o + 2, v.z);
    atomicAdd(o + 3, v.w);
    if (c == 0) atomicAdd(cnt + d, 1.0f);
}

// ---------------------------------------------------------------------------
// WMMA helpers (one wave owns one 16-row tile).
// ---------------------------------------------------------------------------

// Stage 16 x 128 f32 rows -> LDS bf16 tile (row-major, ldk=128).
// scaleByCnt: multiply row i by 1/max(cnt[i],1) (mean finalize).
__device__ __forceinline__ void stage_tile(__bf16* __restrict__ ta,
                                           const float* __restrict__ src,
                                           const float* __restrict__ cnt,
                                           int lane, int rows, int scaleByCnt) {
#pragma unroll
    for (int i = 0; i < 16; ++i) {
        float4 v = make_float4(0.f, 0.f, 0.f, 0.f);
        float r = 1.0f;
        if (i < rows) {
            v = *(const float4*)(src + (size_t)i * D_DIM + lane * 4);
            if (scaleByCnt) r = 1.0f / fmaxf(cnt[i], 1.0f);
        }
        __bf16* d = ta + i * D_DIM + (lane << 2);
        d[0] = (__bf16)(v.x * r);
        d[1] = (__bf16)(v.y * r);
        d[2] = (__bf16)(v.z * r);
        d[3] = (__bf16)(v.w * r);
    }
}

// acc[nt] += A(16xK from LDS bf16, leading dim ldk) @ Wpk(K x NT*16 packed)
template <int KT, int NT>
__device__ __forceinline__ void wave_gemm(const __bf16* __restrict__ tileA, int ldk,
                                          const __bf16* __restrict__ Wpk,
                                          v8f* acc, int lane) {
    const int m = lane & 15;
    const int kh = lane >> 4;
#pragma unroll
    for (int kt = 0; kt < KT; ++kt) {
        const __bf16* p0 = tileA + m * ldk + kt * 32 + kh * 16;
        v8bf lo = *(const v8bf*)p0;
        v8bf hi = *(const v8bf*)(p0 + 8);
        v16bf a = __builtin_shufflevector(lo, hi, 0, 1, 2, 3, 4, 5, 6, 7,
                                          8, 9, 10, 11, 12, 13, 14, 15);
#pragma unroll
        for (int nt = 0; nt < NT; ++nt) {
            v16bf b = *(const v16bf*)(Wpk + ((((kt * NT) + nt) * 32 + lane) << 4));
            acc[nt] = __builtin_amdgcn_wmma_f32_16x16x32_bf16(
                false, a, false, b, (short)0, acc[nt], false, false);
        }
    }
}

// ---------------------------------------------------------------------------
// out_child = relu( x_child @ Wpk_eff + mean_rev @ Wpk_fold + beff + mask*bfold )
// ---------------------------------------------------------------------------
__global__ __launch_bounds__(128) void child_out_kernel(
    const float* __restrict__ x_child,
    const float* __restrict__ sum_rev, const float* __restrict__ cnt_rev,
    const __bf16* __restrict__ Wpk_eff, const __bf16* __restrict__ Wpk_fold,
    const float* __restrict__ beff, const float* __restrict__ bfold,
    float* __restrict__ out, int nNodes) {
    __shared__ __align__(16) __bf16 smemA[4][16 * D_DIM];
    int wave = threadIdx.x >> 5, lane = threadIdx.x & 31;
    int tile = blockIdx.x * 4 + wave;
    int r0 = tile * 16;
    if (r0 >= nNodes) return;
    int rows = nNodes - r0; rows = rows > 16 ? 16 : rows;
    __bf16* ta = smemA[wave];
    const int n = lane & 15, kh = lane >> 4;

    stage_tile(ta, x_child + (size_t)r0 * D_DIM, nullptr, lane, rows, 0);

    v8f acc[8];
#pragma unroll
    for (int nt = 0; nt < 8; ++nt) {
        float bv = beff[nt * 16 + n];
        float bf = bfold[nt * 16 + n];
#pragma unroll
        for (int v = 0; v < 8; ++v) {
            int row = v + (kh << 3);
            int ri = r0 + (row < rows ? row : 0);
            float mk = (cnt_rev[ri] > 0.0f) ? 1.0f : 0.0f;
            acc[nt][v] = bv + mk * bf;
        }
    }
    wave_gemm<4, 8>(ta, D_DIM, Wpk_eff, acc, lane);

    stage_tile(ta, sum_rev + (size_t)r0 * D_DIM, cnt_rev + r0, lane, rows, 1);
    wave_gemm<4, 8>(ta, D_DIM, Wpk_fold, acc, lane);

#pragma unroll
    for (int nt = 0; nt < 8; ++nt)
#pragma unroll
        for (int v = 0; v < 8; ++v) {
            int row = v + (kh << 3);
            if (row < rows)
                out[(size_t)(r0 + row) * D_DIM + nt * 16 + n] = fmaxf(acc[nt][v], 0.0f);
        }
}

// ---------------------------------------------------------------------------
// out_parent = relu( x_p @ Wpk_eff_p + mean_fk @ Wpk_fkfold
//                    + (mask*relu(x_p@W_pred1+b1)) @ Wpk_hfold
//                    + beff_p + mask*bmask_p )
// ---------------------------------------------------------------------------
__global__ __launch_bounds__(128) void parent_out_kernel(
    const float* __restrict__ x_parent,
    const float* __restrict__ sum_fk, const float* __restrict__ cnt_fk,
    const __bf16* __restrict__ Wpk_eff_p, const __bf16* __restrict__ Wpk_pred1,
    const __bf16* __restrict__ Wpk_hfold, const __bf16* __restrict__ Wpk_fkfold,
    const float* __restrict__ b_pred1,
    const float* __restrict__ beff_p, const float* __restrict__ bmask_p,
    float* __restrict__ out, int nNodes) {
    __shared__ __align__(16) __bf16 smemA[4][16 * D_DIM];
    __shared__ __align__(16) __bf16 smemH[4][16 * 64];
    int wave = threadIdx.x >> 5, lane = threadIdx.x & 31;
    int tile = blockIdx.x * 4 + wave;
    int r0 = tile * 16;
    if (r0 >= nNodes) return;
    int rows = nNodes - r0; rows = rows > 16 ? 16 : rows;
    __bf16* ta = smemA[wave];
    __bf16* th = smemH[wave];
    const int n = lane & 15, kh = lane >> 4;

    stage_tile(ta, x_parent + (size_t)r0 * D_DIM, nullptr, lane, rows, 0);

    v8f acc[8];
#pragma unroll
    for (int nt = 0; nt < 8; ++nt) {
        float bv = beff_p[nt * 16 + n];
        float bm = bmask_p[nt * 16 + n];
#pragma unroll
        for (int v = 0; v < 8; ++v) {
            int row = v + (kh << 3);
            int ri = r0 + (row < rows ? row : 0);
            float mk = (cnt_fk[ri] > 0.0f) ? 1.0f : 0.0f;
            acc[nt][v] = bv + mk * bm;
        }
    }
    wave_gemm<4, 8>(ta, D_DIM, Wpk_eff_p, acc, lane);

    // hidden layer of pred MLP: h = mask * relu(x_p @ W_pred1 + b_pred1)
    v8f hacc[4];
#pragma unroll
    for (int nt = 0; nt < 4; ++nt) {
        float bv = b_pred1[nt * 16 + n];
#pragma unroll
        for (int v = 0; v < 8; ++v) hacc[nt][v] = bv;
    }
    wave_gemm<4, 4>(ta, D_DIM, Wpk_pred1, hacc, lane);
#pragma unroll
    for (int nt = 0; nt < 4; ++nt)
#pragma unroll
        for (int v = 0; v < 8; ++v) {
            int row = v + (kh << 3);
            int ri = r0 + (row < rows ? row : 0);
            float mk = (cnt_fk[ri] > 0.0f) ? 1.0f : 0.0f;
            th[row * 64 + nt * 16 + n] = (__bf16)(mk * fmaxf(hacc[nt][v], 0.0f));
        }
    wave_gemm<2, 8>(th, 64, Wpk_hfold, acc, lane);

    stage_tile(ta, sum_fk + (size_t)r0 * D_DIM, cnt_fk + r0, lane, rows, 1);
    wave_gemm<4, 8>(ta, D_DIM, Wpk_fkfold, acc, lane);

#pragma unroll
    for (int nt = 0; nt < 8; ++nt)
#pragma unroll
        for (int v = 0; v < 8; ++v) {
            int row = v + (kh << 3);
            if (row < rows)
                out[(size_t)(r0 + row) * D_DIM + nt * 16 + n] = fmaxf(acc[nt][v], 0.0f);
        }
}

// ---------------------------------------------------------------------------
extern "C" void kernel_launch(void* const* d_in, const int* in_sizes, int n_in,
                              void* d_out, int out_size, void* d_ws, size_t ws_size,
                              hipStream_t stream) {
    (void)n_in; (void)out_size;
    const float* x_child  = (const float*)d_in[0];
    const float* x_parent = (const float*)d_in[1];
    const int*   fk_src   = (const int*)d_in[2];
    const int*   fk_dst   = (const int*)d_in[3];
    const int*   rev_src  = (const int*)d_in[4];
    const int*   rev_dst  = (const int*)d_in[5];
    const float* W_msg_fk = (const float*)d_in[6];
    const float* b_msg_fk = (const float*)d_in[7];
    const float* W_msg_rev = (const float*)d_in[8];
    const float* b_msg_rev = (const float*)d_in[9];
    const float* W_pred1  = (const float*)d_in[10];
    const float* b_pred1  = (const float*)d_in[11];
    const float* W_pred2  = (const float*)d_in[12];
    const float* b_pred2  = (const float*)d_in[13];
    const float* W_self_c = (const float*)d_in[14];
    const float* b_self_c = (const float*)d_in[15];
    const float* W_self_p = (const float*)d_in[16];
    const float* b_self_p = (const float*)d_in[17];
    const float* W_comb_c = (const float*)d_in[18];
    const float* b_comb_c = (const float*)d_in[19];
    const float* W_comb_p = (const float*)d_in[20];
    const float* b_comb_p = (const float*)d_in[21];

    const int NC = in_sizes[0] / D_DIM;
    const int NP = in_sizes[1] / D_DIM;
    const int E  = in_sizes[2];

    // -------- workspace layout --------
    char* w = (char*)d_ws;
    size_t off = 0;
    auto take = [&](size_t bytes) -> char* {
        char* p = w + off;
        off = (off + bytes + 255) & ~(size_t)255;
        return p;
    };
    float*  sum_rev   = (float*)take((size_t)NC * D_DIM * 4);
    float*  sum_fk    = (float*)take((size_t)NP * D_DIM * 4);
    float*  cnt_rev   = (float*)take((size_t)NC * 4);
    float*  cnt_fk    = (float*)take((size_t)NP * 4);
    __bf16* Wpk_eff_c  = (__bf16*)take(128 * 128 * 2);
    __bf16* Wpk_fold_c = (__bf16*)take(128 * 128 * 2);
    __bf16* Wpk_eff_p  = (__bf16*)take(128 * 128 * 2);
    __bf16* Wpk_fkfold = (__bf16*)take(128 * 128 * 2);
    __bf16* Wpk_hfold  = (__bf16*)take(64 * 128 * 2);
    __bf16* Wpk_pred1  = (__bf16*)take(128 * 64 * 2);
    float*  beff_c  = (float*)take(128 * 4);
    float*  bfold_c = (float*)take(128 * 4);
    float*  beff_p  = (float*)take(128 * 4);
    float*  bmask_p = (float*)take(128 * 4);
    if (off > ws_size) return;  // insufficient scratch; refuse to corrupt memory

    hipMemsetAsync(sum_rev, 0, (size_t)NC * D_DIM * 4, stream);
    hipMemsetAsync(sum_fk,  0, (size_t)NP * D_DIM * 4, stream);
    hipMemsetAsync(cnt_rev, 0, (size_t)NC * 4, stream);
    hipMemsetAsync(cnt_fk,  0, (size_t)NP * 4, stream);

    // -------- fold + pack weights (tiny) --------
    const float* WcB = W_comb_c + 128 * 128;
    const float* WpB = W_comb_p + 128 * 128;
    fold_pack_kernel<<<64, 256, 0, stream>>>(W_self_c,  W_comb_c, 1.0f,  0.0f, Wpk_eff_c,  128, 128, 128);
    fold_pack_kernel<<<64, 256, 0, stream>>>(W_msg_rev, WcB,      1.0f,  0.0f, Wpk_fold_c, 128, 128, 128);
    fold_pack_kernel<<<64, 256, 0, stream>>>(W_self_p,  W_comb_p, 1.0f,  0.0f, Wpk_eff_p,  128, 128, 128);
    fold_pack_kernel<<<64, 256, 0, stream>>>(W_msg_fk,  WpB,      0.5f,  0.5f, Wpk_fkfold, 128, 128, 128);
    fold_pack_kernel<<<32, 256, 0, stream>>>(W_pred2,   WpB,     -0.5f,  0.0f, Wpk_hfold,   64, 128, 128);
    fold_pack_kernel<<<32, 256, 0, stream>>>(nullptr,   W_pred1,  0.0f,  1.0f, Wpk_pred1,  128,   0,  64);
    bias_fold_kernel<<<1, 128, 0, stream>>>(b_self_c, W_comb_c, b_comb_c, b_msg_rev,
                                            b_self_p, W_comb_p, b_comb_p, b_msg_fk, b_pred2,
                                            beff_c, bfold_c, beff_p, bmask_p);

    // -------- edge scatter (raw features; means finalized in GEMM kernels) ---
    {
        long long thr = (long long)E * 32;
        int blocks = (int)((thr + 255) / 256);
        scatter_kernel<<<blocks, 256, 0, stream>>>(x_child,  fk_src,  fk_dst,  sum_fk,  cnt_fk,  E);
        scatter_kernel<<<blocks, 256, 0, stream>>>(x_parent, rev_src, rev_dst, sum_rev, cnt_rev, E);
    }

    // -------- fused node-output kernels (WMMA) --------
    float* out_child  = (float*)d_out;
    float* out_parent = (float*)d_out + (size_t)NC * D_DIM;
    int ctiles = (NC + 15) / 16;
    int ptiles = (NP + 15) / 16;
    child_out_kernel<<<(ctiles + 3) / 4, 128, 0, stream>>>(
        x_child, sum_rev, cnt_rev, Wpk_eff_c, Wpk_fold_c, beff_c, bfold_c, out_child, NC);
    parent_out_kernel<<<(ptiles + 3) / 4, 128, 0, stream>>>(
        x_parent, sum_fk, cnt_fk, Wpk_eff_p, Wpk_pred1, Wpk_hfold, Wpk_fkfold,
        b_pred1, beff_p, bmask_p, out_parent, NP);
}
